// LSTM_37245956391346
// MI455X (gfx1250) — matrix operand order repaired and checked by reading, not compiled, not run on hardware
//
#include <hip/hip_runtime.h>
#include <hip/hip_bf16.h>

// Problem constants (match reference).
constexpr int NB = 4096;   // batch
constexpr int NI = 1024;   // input dim
constexpr int NH = 1024;   // hidden dim
constexpr int NO = 1024;   // output dim
constexpr int NG = 4 * NH; // gate width (f,i,o,g concatenated)

typedef __bf16 bf16_t;
typedef __attribute__((ext_vector_type(16))) __bf16 v16bf;
typedef __attribute__((ext_vector_type(8)))  float  v8f;

union FragBf {
    v16bf v;
    uint4 u[2];
};

// gfx1250 async global->LDS staging (ASYNCcnt-tracked DMA into LDS), with a
// compile-safe fallback to VGPR-staged copies if the builtins are absent.
#if defined(__gfx1250__) && \
    __has_builtin(__builtin_amdgcn_global_load_async_to_lds_b128) && \
    __has_builtin(__builtin_amdgcn_s_wait_asynccnt)
#define USE_ASYNC_LDS 1
#define AS1 __attribute__((address_space(1)))
#define AS3 __attribute__((address_space(3)))
typedef int v4i __attribute__((vector_size(16)));   // builtin's pointee type
#else
#define USE_ASYNC_LDS 0
#endif

// ---------------------------------------------------------------------------
// Kernel 1: convert fp32 inputs to bf16 + build concatenated weight/bias bufs
// Launch with exactly NB*NH threads (4M).
// ---------------------------------------------------------------------------
__global__ __launch_bounds__(256) void cvt_pack_kernel(
    const float* __restrict__ x,   const float* __restrict__ h,
    const float* __restrict__ Wxf, const float* __restrict__ Wxi,
    const float* __restrict__ Wxo, const float* __restrict__ Wxg,
    const float* __restrict__ Whf, const float* __restrict__ Whi,
    const float* __restrict__ Who, const float* __restrict__ Whg,
    const float* __restrict__ bxf, const float* __restrict__ bxi,
    const float* __restrict__ bxo, const float* __restrict__ bxg,
    const float* __restrict__ Wout,
    bf16_t* __restrict__ x_b,   bf16_t* __restrict__ h_b,
    bf16_t* __restrict__ Wx_b,  bf16_t* __restrict__ Wh_b,
    bf16_t* __restrict__ Wout_b, float* __restrict__ b_cat)
{
    const int t = blockIdx.x * blockDim.x + threadIdx.x;   // [0, 4M)

    x_b[t] = (bf16_t)x[t];
    h_b[t] = (bf16_t)h[t];

    // Concatenated weights: row n of [4H, K], gate = n>>10, within-gate row = n&1023
    const int n = t >> 10;          // 0..4095
    const int k = t & 1023;
    const int gate = n >> 10;       // 0..3  (f,i,o,g)
    const int r = n & 1023;
    const float* wx = (gate == 0) ? Wxf : (gate == 1) ? Wxi : (gate == 2) ? Wxo : Wxg;
    const float* wh = (gate == 0) ? Whf : (gate == 1) ? Whi : (gate == 2) ? Who : Whg;
    Wx_b[t] = (bf16_t)wx[r * NI + k];
    Wh_b[t] = (bf16_t)wh[r * NH + k];

    if (t < NO * NH)
        Wout_b[t] = (bf16_t)Wout[t];

    if (t < NG) {
        const float* bx = (gate == 0) ? bxf : (gate == 1) ? bxi : (gate == 2) ? bxo : bxg;
        b_cat[t] = bx[r];
    }
}

// ---------------------------------------------------------------------------
// Kernel 2/4: bf16 WMMA GEMM   C[m,n] = sum_k A[m,k]*W[n,k] (+ A1*W1) + bias[n]
// A: [M,K] bf16 row-major, W: [N,K] bf16 row-major (i.e. C = A @ W^T).
// Block: 256 threads (8 wave32), block tile 128(M) x 256(N), K-step 64.
// Wave tile: 64x64 -> 4x4 WMMA 16x16 accumulators. Double-buffered LDS with
// (when available) gfx1250 async global->LDS copies.
// ---------------------------------------------------------------------------
constexpr int BM = 128;
constexpr int BN = 256;
constexpr int BK = 64;
constexpr int LDA = BK + 8;   // padded LDS stride (bf16 elems) -> 144B, 16B aligned

__device__ __forceinline__ void stage_tile(
    const bf16_t* __restrict__ A, const bf16_t* __restrict__ W, int K,
    int m_blk, int n_blk, int kk, int tid,
    bf16_t* As, bf16_t* Bs)
{
#pragma unroll
    for (int i = 0; i < 4; ++i) {
        const int c   = tid + 256 * i;
        const int row = c >> 3;
        const int kc  = (c & 7) * 8;
        const bf16_t* gp = &A[(size_t)(m_blk + row) * K + kk + kc];
        bf16_t* lp = &As[row * LDA + kc];
#if USE_ASYNC_LDS
        __builtin_amdgcn_global_load_async_to_lds_b128(
            (AS1 v4i*)gp, (AS3 v4i*)lp, 0, 0);
#else
        *(uint4*)lp = *(const uint4*)gp;
        if (kk + BK < K) __builtin_prefetch(gp + BK, 0, 3);
#endif
    }
#pragma unroll
    for (int i = 0; i < 8; ++i) {
        const int c   = tid + 256 * i;
        const int row = c >> 3;
        const int kc  = (c & 7) * 8;
        const bf16_t* gp = &W[(size_t)(n_blk + row) * K + kk + kc];
        bf16_t* lp = &Bs[row * LDA + kc];
#if USE_ASYNC_LDS
        __builtin_amdgcn_global_load_async_to_lds_b128(
            (AS1 v4i*)gp, (AS3 v4i*)lp, 0, 0);
#else
        *(uint4*)lp = *(const uint4*)gp;
        if (kk + BK < K) __builtin_prefetch(gp + BK, 0, 3);
#endif
    }
}

__global__ __launch_bounds__(256, 1) void wmma_gemm_kernel(
    const bf16_t* __restrict__ A0, const bf16_t* __restrict__ W0,
    const bf16_t* __restrict__ A1, const bf16_t* __restrict__ W1,
    const float* __restrict__ bias, float* __restrict__ Cout,
    int M, int N, int K, int dualK)
{
    __shared__ bf16_t As[2][BM * LDA];
    __shared__ bf16_t Bs[2][BN * LDA];

    const int tid  = threadIdx.x;
    const int wave = tid >> 5;
    const int lane = tid & 31;
    const int wm   = wave >> 2;      // 0..1
    const int wn   = wave & 3;       // 0..3
    const int lh   = lane >> 4;      // lane half: 0/1
    const int ll   = lane & 15;

    const int m_blk = blockIdx.y * BM;
    const int n_blk = blockIdx.x * BN;

    v8f acc[4][4] = {};

    const int Ktot   = dualK ? 2 * K : K;
    const int ntiles = Ktot / BK;

    // Prologue: issue loads for tile 0 into buffer 0.
    stage_tile(A0, W0, K, m_blk, n_blk, 0, tid, As[0], Bs[0]);

    for (int t = 0; t < ntiles; ++t) {
        const int cur = t & 1;
#if USE_ASYNC_LDS
        __builtin_amdgcn_s_wait_asynccnt(0);   // my async copies for tile t done
#endif
        __syncthreads();                       // everyone's tile t in LDS

        // Issue loads for tile t+1 into the other buffer (overlaps compute).
        if (t + 1 < ntiles) {
            const int kt2 = (t + 1) * BK;
            const bf16_t* A = A0;
            const bf16_t* W = W0;
            int kk = kt2;
            if (dualK && kt2 >= K) { A = A1; W = W1; kk = kt2 - K; }
            stage_tile(A, W, K, m_blk, n_blk, kk, tid, As[cur ^ 1], Bs[cur ^ 1]);
        }

        // Compute tile t.
        const bf16_t* Asb = As[cur];
        const bf16_t* Bsb = Bs[cur];
#pragma unroll
        for (int ks = 0; ks < BK; ks += 32) {
            FragBf afrag[4], bfrag[4];
            // A 16x32 fragment: lanes 0-15 row m, K {0-7,16-23}; lanes 16-31 K {8-15,24-31}
#pragma unroll
            for (int i = 0; i < 4; ++i) {
                const int row = wm * 64 + i * 16 + ll;
                const bf16_t* p = &Asb[row * LDA + ks + lh * 8];
                afrag[i].u[0] = *(const uint4*)p;
                afrag[i].u[1] = *(const uint4*)(p + 16);
            }
            // B 32x16 fragment: lane n holds column n; lanes 0-15 K {0-7,16-23}, lanes 16-31 K {8-15,24-31}
#pragma unroll
            for (int j = 0; j < 4; ++j) {
                const int col = wn * 64 + j * 16 + ll;
                const bf16_t* p = &Bsb[col * LDA + ks + lh * 8];
                bfrag[j].u[0] = *(const uint4*)p;
                bfrag[j].u[1] = *(const uint4*)(p + 16);
            }
#pragma unroll
            for (int i = 0; i < 4; ++i)
#pragma unroll
                for (int j = 0; j < 4; ++j)
                    acc[i][j] = __builtin_amdgcn_wmma_f32_16x16x32_bf16(
                        false, afrag[i].v, false, bfrag[j].v,
                        (short)0, acc[i][j], false, false);
        }
        __syncthreads();   // tile t fully consumed before buffer reuse at t+2
    }

    // Epilogue: C/D layout -> element r of v8f is row (r + 8*lh), col = ll
#pragma unroll
    for (int i = 0; i < 4; ++i) {
#pragma unroll
        for (int j = 0; j < 4; ++j) {
            const int col = n_blk + wn * 64 + j * 16 + ll;
            const float b = bias[col];
#pragma unroll
            for (int r = 0; r < 8; ++r) {
                const int row = m_blk + wm * 64 + i * 16 + r + 8 * lh;
                Cout[(size_t)row * N + col] = acc[i][j][r] + b;
            }
        }
    }
}

// ---------------------------------------------------------------------------
// Kernel 3: LSTM elementwise. gates already include bias.
// ---------------------------------------------------------------------------
__global__ __launch_bounds__(256) void lstm_eltwise_kernel(
    const float* __restrict__ gates, const float* __restrict__ cell,
    float* __restrict__ h_out, float* __restrict__ c_out,
    bf16_t* __restrict__ h_b)
{
    const int t = blockIdx.x * blockDim.x + threadIdx.x;   // [0, 4M)
    const int m = t >> 10;
    const int n = t & 1023;
    const float* gm = gates + (size_t)m * NG;

    const float f = 1.0f / (1.0f + __expf(-gm[n]));
    const float i = 1.0f / (1.0f + __expf(-gm[n + NH]));
    const float o = 1.0f / (1.0f + __expf(-gm[n + 2 * NH]));
    const float g = tanhf(gm[n + 3 * NH]);

    const float cn = f * cell[t] + i * g;
    const float hn = o * tanhf(cn);

    c_out[t] = cn;
    h_out[t] = hn;
    h_b[t]   = (bf16_t)hn;
}

// ---------------------------------------------------------------------------
extern "C" void kernel_launch(void* const* d_in, const int* in_sizes, int n_in,
                              void* d_out, int out_size, void* d_ws, size_t ws_size,
                              hipStream_t stream)
{
    const float* x    = (const float*)d_in[0];
    const float* h    = (const float*)d_in[1];
    const float* cell = (const float*)d_in[2];
    const float* Wxf = (const float*)d_in[3];
    const float* bxf = (const float*)d_in[4];
    const float* Whf = (const float*)d_in[5];
    const float* Wxi = (const float*)d_in[6];
    const float* bxi = (const float*)d_in[7];
    const float* Whi = (const float*)d_in[8];
    const float* Wxo = (const float*)d_in[9];
    const float* bxo = (const float*)d_in[10];
    const float* Who = (const float*)d_in[11];
    const float* Wxg = (const float*)d_in[12];
    const float* bxg = (const float*)d_in[13];
    const float* Whg = (const float*)d_in[14];
    const float* Wout = (const float*)d_in[15];
    const float* bout = (const float*)d_in[16];

    float* out_h = (float*)d_out;                 // new_hidden [4096,1024]
    float* out_c = out_h + (size_t)NB * NH;       // new_cell   [4096,1024]
    float* out_v = out_c + (size_t)NB * NH;       // output_vec [4096,1024]

    // Workspace carving (all chunks are multiples of 16B; base is 256B aligned)
    char* w = (char*)d_ws;
    bf16_t* x_b    = (bf16_t*)w; w += (size_t)NB * NI * sizeof(bf16_t);   // 8 MB
    bf16_t* h_b    = (bf16_t*)w; w += (size_t)NB * NH * sizeof(bf16_t);   // 8 MB
    bf16_t* Wx_b   = (bf16_t*)w; w += (size_t)NG * NI * sizeof(bf16_t);   // 8 MB
    bf16_t* Wh_b   = (bf16_t*)w; w += (size_t)NG * NH * sizeof(bf16_t);   // 8 MB
    bf16_t* Wout_b = (bf16_t*)w; w += (size_t)NO * NH * sizeof(bf16_t);   // 2 MB
    bf16_t* hnew_b = (bf16_t*)w; w += (size_t)NB * NH * sizeof(bf16_t);   // 8 MB
    float*  b_cat  = (float*)w;  w += (size_t)NG * sizeof(float);         // 16 KB
    float*  gates  = (float*)w;  w += (size_t)NB * NG * sizeof(float);    // 64 MB

    const int n_elems = NB * NH;                  // 4M
    const int cvt_blocks = n_elems / 256;

    // 1) convert/pack
    cvt_pack_kernel<<<cvt_blocks, 256, 0, stream>>>(
        x, h, Wxf, Wxi, Wxo, Wxg, Whf, Whi, Who, Whg,
        bxf, bxi, bxo, bxg, Wout,
        x_b, h_b, Wx_b, Wh_b, Wout_b, b_cat);

    // 2) gates = x @ Wx^T + h @ Wh^T + b   -> [4096, 4096]
    dim3 gridGates(NG / BN, NB / BM);             // (16, 32)
    wmma_gemm_kernel<<<gridGates, 256, 0, stream>>>(
        x_b, Wx_b, h_b, Wh_b, b_cat, gates, NB, NG, NI, 1);

    // 3) elementwise LSTM
    lstm_eltwise_kernel<<<cvt_blocks, 256, 0, stream>>>(
        gates, cell, out_h, out_c, hnew_b);

    // 4) output_vec = new_hidden @ W_out^T + b_out  -> [4096, 1024]
    dim3 gridOut(NO / BN, NB / BM);               // (4, 32)
    wmma_gemm_kernel<<<gridOut, 256, 0, stream>>>(
        hnew_b, Wout_b, (const bf16_t*)nullptr, (const bf16_t*)nullptr,
        bout, out_v, NB, NO, NH, 0);
}
